// EngramMHFFNOBlock_38199439131342
// MI455X (gfx1250) — matrix operand order repaired and checked by reading, not compiled, not run on hardware
//
#include <hip/hip_runtime.h>
#include <hip/hip_fp16.h>
#include <math.h>

typedef __attribute__((ext_vector_type(16))) _Float16 v16h;
typedef __attribute__((ext_vector_type(8)))  _Float16 v8h;
typedef __attribute__((ext_vector_type(8)))  float    v8f;

#define TWO_PI_OVER_128 0.04908738521234052f

__device__ __forceinline__ float gelu_exact(float v) {
  return 0.5f * v * (1.0f + erff(v * 0.7071067811865476f));
}

// ---- WMMA fragment helpers (wave32, A/B in contiguous k order k = 16*g + j; C/D per CDNA5 ISA) ----
// A matrix 16x32 f16, row-major in LDS, leading dim lda (halves). Per-lane data
// (k = j + 16*g at fixed m) is 16 contiguous halves -> one 32B load.
__device__ __forceinline__ v16h frag_a_lds(const _Float16* base, int lda) {
  const int lane = threadIdx.x & 31;
  const int m = lane & 15, g = lane >> 4;
  return *(const v16h*)(base + m * lda + g * 16);
}
// B matrix 32x16 (K x N) with LDS holding B TRANSPOSED: Bt[n][k], leading dim ldt.
// Per-lane data (k = j + 16*g at fixed n) is 16 contiguous halves -> one 32B load.
__device__ __forceinline__ v16h frag_b_t(const _Float16* baseT, int ldt) {
  const int lane = threadIdx.x & 31;
  const int n = lane & 15, g = lane >> 4;
  return *(const v16h*)(baseT + n * ldt + g * 16);
}
// B matrix 32x16 loaded directly from global f32 row-major (K x N), converted to f16.
__device__ __forceinline__ v16h frag_b_gf32(const float* base, int ldb) {
  const int lane = threadIdx.x & 31;
  const int n = lane & 15, g = lane >> 4;
  v16h b;
#pragma unroll
  for (int j = 0; j < 16; ++j) b[j] = (_Float16)base[(j + g * 16) * ldb + n];
  return b;
}
__device__ __forceinline__ v8f wmma16(v16h a, v16h b, v8f c) {
  return __builtin_amdgcn_wmma_f32_16x16x32_f16(false, a, false, b, (short)0, c, false, false);
}
// C/D element (r, lane): m = r + 8*(lane>>4), n = lane&15

// -------------------------------------------------------------------------
// Kernel 0: zero small accumulator buffers (atomics targets)
__global__ void k_init(float* magb, float* bstat) {
  const int t = threadIdx.x;
  if (t < 8)  magb[t]  = 0.0f;
  if (t < 16) bstat[t] = 0.0f;
}

// -------------------------------------------------------------------------
// Kernel 1: proj[10000,64] = se_emb[10000,64] @ se_w[64,64] + se_b  (WMMA)
__global__ void __launch_bounds__(128) k_se_proj(const float* __restrict__ se_emb,
                                                 const float* __restrict__ se_w,
                                                 const float* __restrict__ se_b,
                                                 float* __restrict__ proj) {
  __shared__ __align__(32) _Float16 Aemb[16][64];
  __shared__ __align__(32) _Float16 WseT[64][64];  // [n=cout][k=cin]
  const int mt = blockIdx.x, t = threadIdx.x, wave = t >> 5;
  for (int i = t; i < 16 * 64; i += 128)
    Aemb[i >> 6][i & 63] = (_Float16)se_emb[(size_t)mt * 1024 + i];
  for (int i = t; i < 64 * 64; i += 128)
    WseT[i & 63][i >> 6] = (_Float16)se_w[i];   // se_w[k*64+n] -> WseT[n][k]
  __syncthreads();
  const int nt = wave;  // 4 waves <-> 4 N-tiles
  v8f acc = {};
#pragma unroll
  for (int kk = 0; kk < 2; ++kk)
    acc = wmma16(frag_a_lds(&Aemb[0][kk * 32], 64), frag_b_t(&WseT[nt * 16][kk * 32], 64), acc);
  const int lane = t & 31, n = lane & 15, g = lane >> 4;
  const int col = nt * 16 + n;
#pragma unroll
  for (int e = 0; e < 8; ++e)
    proj[(size_t)(mt * 16 + e + 8 * g) * 64 + col] = acc[e] + se_b[col];
}

// -------------------------------------------------------------------------
// Kernel 2: per-pixel 3x3 hash -> idx -> gather proj row into spatial(B,C,H,W)
__global__ void __launch_bounds__(256) k_hash(const float* __restrict__ x,
                                              const float* __restrict__ proj,
                                              float* __restrict__ spat) {
  const int p = blockIdx.x * 256 + threadIdx.x;   // B*H*W = 131072
  const int b = p >> 14, rem = p & 16383;
  const int h = rem >> 7, w = rem & 127;
  float accf = 0.0f;
  for (int c = 0; c < 64; ++c) {
    const float* xc = x + ((size_t)(b * 64 + c) << 14);
    int s = 0;
#pragma unroll
    for (int di = -1; di <= 1; ++di) {
      const int hh = min(max(h + di, 0), 127);
#pragma unroll
      for (int dj = -1; dj <= 1; ++dj) {
        const int ww = min(max(w + dj, 0), 127);
        s += (int)(xc[hh * 128 + ww] * 100.0f);   // trunc toward zero == astype(int32)
      }
    }
    const int m = (s < 0 ? -s : s) % 10000;
    accf += (float)m;
  }
  const int idx = (int)(accf * (1.0f / 64.0f));
  __builtin_prefetch(&proj[(size_t)idx * 64], 0, 3);  // global_prefetch_b8
  for (int c = 0; c < 64; ++c)
    spat[((size_t)(b * 64 + c) << 14) + rem] = proj[(size_t)idx * 64 + c];
}

// -------------------------------------------------------------------------
// Kernel 3: low-band forward DFT per (b,c) image: lo_r/lo_i (32x17), + |lo| sums
__global__ void __launch_bounds__(256) k_lo(const float* __restrict__ x,
                                            float* __restrict__ lo_r,
                                            float* __restrict__ lo_i,
                                            float* __restrict__ magb) {
  __shared__ __align__(32) _Float16 cosA[32][128];    // (X, h)  A operand
  __shared__ __align__(32) _Float16 sinA[32][128];
  __shared__ __align__(32) _Float16 Umat[32][128];    // (X, w)  A operand of stage 2
  __shared__ __align__(32) _Float16 Vmat[32][128];
  __shared__ __align__(32) _Float16 cosBt2[32][128];  // Bt: [n=Y][k=w]
  __shared__ __align__(32) _Float16 sinBt2[32][128];
  const int bc = blockIdx.x, t = threadIdx.x, wave = t >> 5;
  const float* xg = x + ((size_t)bc << 14);

  for (int i = t; i < 32 * 128; i += 256) {
    const int X = i >> 7, h = i & 127;
    float s, c;
    __sincosf((float)((X * h) & 127) * TWO_PI_OVER_128, &s, &c);
    cosA[X][h] = (_Float16)c;
    sinA[X][h] = (_Float16)s;
    cosBt2[X][h] = (_Float16)c;   // same function of (Y=X, w=h)
    sinBt2[X][h] = (_Float16)s;
  }
  __syncthreads();

  // Stage 1: U = cosA @ img, V = sinA @ img  (M=32,K=128,N=128)
  // wave <-> N-tile; B fragment loaded once per k-step, 4 accumulators.
  {
    const int Nt = wave;  // 0..7
    v8f aU0 = {}, aU1 = {}, aV0 = {}, aV1 = {};
#pragma unroll
    for (int kk = 0; kk < 4; ++kk) {
      const v16h b = frag_b_gf32(xg + (kk * 32) * 128 + Nt * 16, 128);
      const v16h c0 = frag_a_lds(&cosA[0][kk * 32], 128);
      const v16h c1 = frag_a_lds(&cosA[16][kk * 32], 128);
      const v16h s0 = frag_a_lds(&sinA[0][kk * 32], 128);
      const v16h s1 = frag_a_lds(&sinA[16][kk * 32], 128);
      aU0 = wmma16(c0, b, aU0);
      aU1 = wmma16(c1, b, aU1);
      aV0 = wmma16(s0, b, aV0);
      aV1 = wmma16(s1, b, aV1);
    }
    const int lane = t & 31, n = lane & 15, g = lane >> 4;
#pragma unroll
    for (int e = 0; e < 8; ++e) {
      Umat[e + 8 * g][Nt * 16 + n]      = (_Float16)aU0[e];
      Umat[16 + e + 8 * g][Nt * 16 + n] = (_Float16)aU1[e];
      Vmat[e + 8 * g][Nt * 16 + n]      = (_Float16)aV0[e];
      Vmat[16 + e + 8 * g][Nt * 16 + n] = (_Float16)aV1[e];
    }
  }
  __syncthreads();

  // Stage 2: lo_r = (U@cosB - V@sinB)/128 ; lo_i = -(U@sinB + V@cosB)/128
  if (wave < 4) {
    const int Mt = wave >> 1, Nt = wave & 1;
    v8f aR1 = {}, aR2 = {}, aI1 = {}, aI2 = {};
#pragma unroll
    for (int kk = 0; kk < 4; ++kk) {
      const v16h aU = frag_a_lds(&Umat[Mt * 16][kk * 32], 128);
      const v16h aV = frag_a_lds(&Vmat[Mt * 16][kk * 32], 128);
      const v16h bC = frag_b_t(&cosBt2[Nt * 16][kk * 32], 128);
      const v16h bS = frag_b_t(&sinBt2[Nt * 16][kk * 32], 128);
      aR1 = wmma16(aU, bC, aR1);
      aR2 = wmma16(aV, bS, aR2);
      aI1 = wmma16(aU, bS, aI1);
      aI2 = wmma16(aV, bC, aI2);
    }
    const int lane = t & 31, n = lane & 15, g = lane >> 4;
    const int Y = Nt * 16 + n;
    float msum = 0.0f;
#pragma unroll
    for (int e = 0; e < 8; ++e) {
      const int X = Mt * 16 + e + 8 * g;
      const float r  = (aR1[e] - aR2[e]) * (1.0f / 128.0f);
      const float im = -(aI1[e] + aI2[e]) * (1.0f / 128.0f);
      if (Y < 17) {
        lo_r[(size_t)bc * 544 + X * 17 + Y] = r;
        lo_i[(size_t)bc * 544 + X * 17 + Y] = im;
        msum += sqrtf(r * r + im * im);
      }
    }
    for (int off = 16; off > 0; off >>= 1) msum += __shfl_down(msum, off, 32);
    if (lane == 0) atomicAdd(&magb[bc >> 6], msum);
  }
}

// -------------------------------------------------------------------------
// Kernel 4: of[b, h*16+o, X, Y] = sum_i lo[b,i]*wc[h,i,o]  (complex) per freq point
__global__ void __launch_bounds__(512) k_einsum(const float* __restrict__ lo_r,
                                                const float* __restrict__ lo_i,
                                                const float* __restrict__ wr,
                                                const float* __restrict__ wi,
                                                float* __restrict__ ofr,
                                                float* __restrict__ ofi) {
  __shared__ float Lr[8][64];
  __shared__ float Li[8][64];
  const int f = blockIdx.x, t = threadIdx.x;
  {
    const int bb = t >> 6, ii = t & 63;
    Lr[bb][ii] = lo_r[(size_t)(bb * 64 + ii) * 544 + f];
    Li[bb][ii] = lo_i[(size_t)(bb * 64 + ii) * 544 + f];
  }
  __syncthreads();
  const int b = t >> 6, co = t & 63, hh = co >> 4, o = co & 15;
  const size_t wbase = ((size_t)(hh * 64) * 16 + o) * 544 + f;
  __builtin_prefetch(&wr[wbase], 0, 3);
  __builtin_prefetch(&wi[wbase], 0, 3);
  float orr = 0.0f, oii = 0.0f;
  for (int i = 0; i < 64; ++i) {
    const size_t widx = wbase + (size_t)i * (16 * 544);
    const float a = wr[widx], bb2 = wi[widx];
    const float lr = Lr[b][i], li = Li[b][i];
    orr += lr * a - li * bb2;
    oii += lr * bb2 + li * a;
  }
  ofr[(size_t)(b * 64 + co) * 544 + f] = orr;
  ofi[(size_t)(b * 64 + co) * 544 + f] = oii;
}

// -------------------------------------------------------------------------
// Kernel 5a: fidx[b] = int(mag[b]*1000) % 5000
__global__ void k_fidx(const float* magb, int* fidxb) {
  const int t = threadIdx.x;
  if (t < 8) {
    const float mag = magb[t] / 34816.0f;   // mean over C*MX*MY
    fidxb[t] = ((int)(mag * 1000.0f)) % 5000;
  }
}

// Kernel 5b: re/im[b, c*544+f] = sp_emb[fidx[b]] @ sp_wr/wi + bias
__global__ void __launch_bounds__(256) k_specfreq(const int* __restrict__ fidxb,
                                                  const float* __restrict__ sp_emb,
                                                  const float* __restrict__ sp_wr,
                                                  const float* __restrict__ sp_br,
                                                  const float* __restrict__ sp_wi,
                                                  const float* __restrict__ sp_bi,
                                                  float* __restrict__ re,
                                                  float* __restrict__ im) {
  const int j = blockIdx.x * 256 + threadIdx.x;  // 8*34816 = 278528
  const int b = j / 34816, jj = j % 34816;
  const float* fe = sp_emb + (size_t)fidxb[b] * 32;
  float r = sp_br[jj], q = sp_bi[jj];
  for (int e = 0; e < 32; ++e) {
    const float f = fe[e];
    r += f * sp_wr[(size_t)e * 34816 + jj];
    q += f * sp_wi[(size_t)e * 34816 + jj];
  }
  re[j] = r;
  im[j] = q;
}

// -------------------------------------------------------------------------
// Kernel 6: irfft2 of 32x17 low band -> 128x128 image, WMMA two-stage
__global__ void __launch_bounds__(256) k_irfft(const float* __restrict__ Fr,
                                               const float* __restrict__ Fi,
                                               float* __restrict__ out,
                                               const float* __restrict__ bias,
                                               int has_bias) {
  __shared__ __align__(32) _Float16 FrA[32][32];     // (X, Y) cY folded, zero padded
  __shared__ __align__(32) _Float16 FiA[32][32];
  __shared__ __align__(32) _Float16 cBT[128][32];    // Bt of stage1: [n=w][k=Y]
  __shared__ __align__(32) _Float16 sBT[128][32];
  __shared__ __align__(32) _Float16 PmT[128][32];    // Bt of stage2: [n=w][k=X]
  __shared__ __align__(32) _Float16 QmT[128][32];
  __shared__ __align__(32) _Float16 cosA2[128][32];  // (h, X) A operand
  __shared__ __align__(32) _Float16 sinA2[128][32];
  const int bc = blockIdx.x, t = threadIdx.x, wave = t >> 5;

  for (int i = t; i < 32 * 32; i += 256) {
    const int X = i >> 5, Y = i & 31;
    float fr = 0.0f, fi = 0.0f;
    if (Y < 17) {
      const float cY = (Y == 0) ? 1.0f : 2.0f;
      fr = cY * Fr[(size_t)bc * 544 + X * 17 + Y];
      fi = cY * Fi[(size_t)bc * 544 + X * 17 + Y];
    }
    FrA[X][Y] = (_Float16)fr;
    FiA[X][Y] = (_Float16)fi;
  }
  for (int i = t; i < 128 * 32; i += 256) {
    const int a = i >> 5, k = i & 31;   // a: w or h ; k: Y or X
    float s, c;
    __sincosf((float)((k * a) & 127) * TWO_PI_OVER_128, &s, &c);
    cBT[a][k] = (_Float16)c;
    sBT[a][k] = (_Float16)s;
    cosA2[a][k] = (_Float16)c;
    sinA2[a][k] = (_Float16)s;
  }
  __syncthreads();

  // Stage 1: P = FrA@cosB - FiA@sinB ; Q = FrA@sinB + FiA@cosB  (M=32,K=32,N=128)
  // wave <-> N-tile; covers both M-tiles with shared B fragments.
  {
    const int Nt = wave;  // 0..7
    const v16h bC = frag_b_t(&cBT[Nt * 16][0], 32);
    const v16h bS = frag_b_t(&sBT[Nt * 16][0], 32);
    const int lane = t & 31, n = lane & 15, g = lane >> 4;
#pragma unroll
    for (int Mt = 0; Mt < 2; ++Mt) {
      const v16h aR = frag_a_lds(&FrA[Mt * 16][0], 32);
      const v16h aI = frag_a_lds(&FiA[Mt * 16][0], 32);
      v8f p1 = {}, p2 = {}, q1 = {}, q2 = {};
      p1 = wmma16(aR, bC, p1);
      p2 = wmma16(aI, bS, p2);
      q1 = wmma16(aR, bS, q1);
      q2 = wmma16(aI, bC, q2);
#pragma unroll
      for (int e = 0; e < 8; ++e) {
        PmT[Nt * 16 + n][Mt * 16 + e + 8 * g] = (_Float16)(p1[e] - p2[e]);
        QmT[Nt * 16 + n][Mt * 16 + e + 8 * g] = (_Float16)(q1[e] + q2[e]);
      }
    }
  }
  __syncthreads();

  // Stage 2: y = (cosA2@P - sinA2@Q)/128 (+bias)  (M=128,K=32,N=128)
  // wave <-> N-tile; B fragments loaded once, loop over 8 M-tiles.
  {
    const int Nt = wave;
    const v16h bP = frag_b_t(&PmT[Nt * 16][0], 32);
    const v16h bQ = frag_b_t(&QmT[Nt * 16][0], 32);
    const float bv = has_bias ? bias[bc & 63] : 0.0f;
    const int lane = t & 31, n = lane & 15, g = lane >> 4;
#pragma unroll
    for (int Mt = 0; Mt < 8; ++Mt) {
      const v16h aC = frag_a_lds(&cosA2[Mt * 16][0], 32);
      const v16h aS = frag_a_lds(&sinA2[Mt * 16][0], 32);
      v8f y1 = {}, y2 = {};
      y1 = wmma16(aC, bP, y1);
      y2 = wmma16(aS, bQ, y2);
#pragma unroll
      for (int e = 0; e < 8; ++e) {
        const int hh = Mt * 16 + e + 8 * g, ww = Nt * 16 + n;
        out[((size_t)bc << 14) + hh * 128 + ww] = (y1[e] - y2[e]) * (1.0f / 128.0f) + bv;
      }
    }
  }
}

// -------------------------------------------------------------------------
// Kernel 7: per-plane mean (B*C planes of 16384)
__global__ void __launch_bounds__(256) k_mean(const float* __restrict__ in,
                                              float* __restrict__ outm) {
  __shared__ float red[256];
  const int p = blockIdx.x, t = threadIdx.x;
  const float* base = in + ((size_t)p << 14);
  float s = 0.0f;
  for (int i = t; i < 16384; i += 256) s += base[i];
  red[t] = s;
  __syncthreads();
  for (int st = 128; st > 0; st >>= 1) {
    if (t < st) red[t] += red[t + st];
    __syncthreads();
  }
  if (t == 0) outm[p] = red[0] / 16384.0f;
}

// -------------------------------------------------------------------------
// Kernel 8: gate MLP + softmax -> gw[B,3]
__global__ void __launch_bounds__(512) k_gate(const float* __restrict__ m_sp,
                                              const float* __restrict__ m_mhf,
                                              const float* __restrict__ m_spec,
                                              const float* __restrict__ w1,
                                              const float* __restrict__ b1,
                                              const float* __restrict__ w2,
                                              const float* __restrict__ b2,
                                              float* __restrict__ gwb) {
  __shared__ float gin[8][192];
  __shared__ float z1[8][64];
  __shared__ float z2[8][3];
  const int t = threadIdx.x;
  for (int i = t; i < 8 * 192; i += 512) {
    const int bb = i / 192, k = i % 192;
    gin[bb][k] = (k < 64) ? m_sp[bb * 64 + k]
               : (k < 128) ? m_mhf[bb * 64 + (k - 64)]
                           : m_spec[bb * 64 + (k - 128)];
  }
  __syncthreads();
  const int b = t >> 6, j = t & 63;
  float z = b1[j];
  for (int k = 0; k < 192; ++k) z += gin[b][k] * w1[k * 64 + j];
  z1[b][j] = gelu_exact(z);
  __syncthreads();
  if (j < 3) {
    float s = b2[j];
    for (int k = 0; k < 64; ++k) s += z1[b][k] * w2[k * 3 + j];
    z2[b][j] = s;
  }
  __syncthreads();
  if (j == 0) {
    const float m = fmaxf(z2[b][0], fmaxf(z2[b][1], z2[b][2]));
    const float e0 = expf(z2[b][0] - m), e1 = expf(z2[b][1] - m), e2 = expf(z2[b][2] - m);
    const float inv = 1.0f / (e0 + e1 + e2);
    gwb[b * 3 + 0] = e0 * inv;
    gwb[b * 3 + 1] = e1 * inv;
    gwb[b * 3 + 2] = e2 * inv;
  }
}

// -------------------------------------------------------------------------
// Kernel 9: y = a*spatial + b*mhf + g*spec + x@skip_w + skip_b; batch sum/sumsq
__global__ void __launch_bounds__(256) k_y(const float* __restrict__ x,
                                           const float* __restrict__ skip_w,
                                           const float* __restrict__ skip_b,
                                           const float* __restrict__ spat,
                                           const float* __restrict__ mhf,
                                           const float* __restrict__ spec,
                                           const float* __restrict__ gw,
                                           float* __restrict__ yout,
                                           float* __restrict__ bstat) {
  __shared__ __align__(32) _Float16 Axt[128][64];   // pixel w x cin (A operand)
  __shared__ __align__(32) _Float16 WskT[64][64];   // Bt: [n=cout][k=cin]
  const int bh = blockIdx.x, b = bh >> 7, h = bh & 127, t = threadIdx.x, wave = t >> 5;
  for (int i = t; i < 128 * 64; i += 256) {
    const int c = i >> 7, w = i & 127;
    Axt[w][c] = (_Float16)x[((size_t)(b * 64 + c) << 14) + (h << 7) + w];
  }
  for (int i = t; i < 64 * 64; i += 256) WskT[i & 63][i >> 6] = (_Float16)skip_w[i];
  __syncthreads();
  const float ga = gw[b * 3], gb2 = gw[b * 3 + 1], gc = gw[b * 3 + 2];
  float ls = 0.0f, ls2 = 0.0f;
  // wave <-> (Nt = wave&3, Mhalf = wave>>2): 4 M-tiles per wave, B frag reused.
  {
    const int Nt = wave & 3, Mh = wave >> 2;
    v8f acc[4] = {};
#pragma unroll
    for (int kk = 0; kk < 2; ++kk) {
      const v16h bW = frag_b_t(&WskT[Nt * 16][kk * 32], 64);
#pragma unroll
      for (int ml = 0; ml < 4; ++ml)
        acc[ml] = wmma16(frag_a_lds(&Axt[(Mh * 4 + ml) * 16][kk * 32], 64), bW, acc[ml]);
    }
    const int lane = t & 31, n = lane & 15, g = lane >> 4;
    const int cout = Nt * 16 + n;
    const float sb = skip_b[cout];
#pragma unroll
    for (int ml = 0; ml < 4; ++ml) {
#pragma unroll
      for (int e = 0; e < 8; ++e) {
        const int w = (Mh * 4 + ml) * 16 + e + 8 * g;
        const size_t pix = ((size_t)(b * 64 + cout) << 14) + (h << 7) + w;
        const float v = acc[ml][e] + sb + ga * spat[pix] + gb2 * mhf[pix] + gc * spec[pix];
        yout[pix] = v;
        ls += v;
        ls2 += v * v;
      }
    }
  }
  for (int off = 16; off > 0; off >>= 1) {
    ls += __shfl_down(ls, off, 32);
    ls2 += __shfl_down(ls2, off, 32);
  }
  if ((t & 31) == 0) {
    atomicAdd(&bstat[b * 2 + 0], ls);
    atomicAdd(&bstat[b * 2 + 1], ls2);
  }
}

// -------------------------------------------------------------------------
// Kernel 10: layernorm over (C,H,W) per batch; MLP 64->128 gelu ->64; out = y + h
__global__ void __launch_bounds__(256) k_norm_mlp(float* __restrict__ yio,
                                                  const float* __restrict__ bstat,
                                                  const float* __restrict__ norm_g,
                                                  const float* __restrict__ norm_b,
                                                  const float* __restrict__ mlp_w1,
                                                  const float* __restrict__ mlp_b1,
                                                  const float* __restrict__ mlp_w2,
                                                  const float* __restrict__ mlp_b2) {
  __shared__ __align__(32) _Float16 Ay[128][64];
  __shared__ __align__(32) _Float16 H1[128][128];
  __shared__ __align__(32) _Float16 WT[8192];  // union: W1T[128][64] then W2T[64][128]
  const int bh = blockIdx.x, b = bh >> 7, h = bh & 127, t = threadIdx.x, wave = t >> 5;
  const float Np = 1048576.0f;  // C*H*W
  const float mu = bstat[b * 2] / Np;
  const float var = bstat[b * 2 + 1] / Np - mu * mu;
  const float rstd = rsqrtf(var + 1e-5f);
  __builtin_prefetch(&mlp_w2[(size_t)t * 32], 0, 3);  // warm L2 for phase 2
  for (int i = t; i < 128 * 64; i += 256) {
    const int c = i >> 7, w = i & 127;
    const float yv = yio[((size_t)(b * 64 + c) << 14) + (h << 7) + w];
    Ay[w][c] = (_Float16)((yv - mu) * rstd * norm_g[c] + norm_b[c]);
  }
  for (int i = t; i < 64 * 128; i += 256)
    WT[(i & 127) * 64 + (i >> 7)] = (_Float16)mlp_w1[i];  // mlp_w1[k*128+n] -> W1T[n][k]
  __syncthreads();
  // GEMM1: M=128(pixels) K=64 N=128; wave <-> N-tile, 8 M accumulators.
  {
    const int Nt = wave;
    v8f acc[8] = {};
#pragma unroll
    for (int kk = 0; kk < 2; ++kk) {
      const v16h bW = frag_b_t(&WT[(Nt * 16) * 64 + kk * 32], 64);
#pragma unroll
      for (int Mt = 0; Mt < 8; ++Mt)
        acc[Mt] = wmma16(frag_a_lds(&Ay[Mt * 16][kk * 32], 64), bW, acc[Mt]);
    }
    const int lane = t & 31, n = lane & 15, g = lane >> 4;
    const int j = Nt * 16 + n;
    const float b1 = mlp_b1[j];
#pragma unroll
    for (int Mt = 0; Mt < 8; ++Mt)
#pragma unroll
      for (int e = 0; e < 8; ++e)
        H1[Mt * 16 + e + 8 * g][j] = (_Float16)gelu_exact(acc[Mt][e] + b1);
  }
  __syncthreads();
  for (int i = t; i < 128 * 64; i += 256)
    WT[(i & 63) * 128 + (i >> 6)] = (_Float16)mlp_w2[i];  // mlp_w2[k*64+n] -> W2T[n][k]
  __syncthreads();
  // GEMM2: M=128 K=128 N=64; wave <-> (Nt, Mhalf); final = y + h
  {
    const int Nt = wave & 3, Mh = wave >> 2;
    v8f acc[4] = {};
#pragma unroll
    for (int kk = 0; kk < 4; ++kk) {
      const v16h bW = frag_b_t(&WT[(Nt * 16) * 128 + kk * 32], 128);
#pragma unroll
      for (int ml = 0; ml < 4; ++ml)
        acc[ml] = wmma16(frag_a_lds(&H1[(Mh * 4 + ml) * 16][kk * 32], 128), bW, acc[ml]);
    }
    const int lane = t & 31, n = lane & 15, g = lane >> 4;
    const int cout = Nt * 16 + n;
    const float b2 = mlp_b2[cout];
#pragma unroll
    for (int ml = 0; ml < 4; ++ml) {
#pragma unroll
      for (int e = 0; e < 8; ++e) {
        const int w = (Mh * 4 + ml) * 16 + e + 8 * g;
        const size_t pix = ((size_t)(b * 64 + cout) << 14) + (h << 7) + w;
        yio[pix] = yio[pix] + acc[ml][e] + b2;
      }
    }
  }
}

// -------------------------------------------------------------------------
extern "C" void kernel_launch(void* const* d_in, const int* in_sizes, int n_in,
                              void* d_out, int out_size, void* d_ws, size_t ws_size,
                              hipStream_t stream) {
  (void)in_sizes; (void)n_in; (void)out_size; (void)ws_size;
  const float* x        = (const float*)d_in[0];
  const float* se_emb   = (const float*)d_in[1];
  const float* se_w     = (const float*)d_in[2];
  const float* se_b     = (const float*)d_in[3];
  const float* mhf_wr   = (const float*)d_in[4];
  const float* mhf_wi   = (const float*)d_in[5];
  const float* mhf_bias = (const float*)d_in[6];
  const float* sp_emb   = (const float*)d_in[7];
  const float* sp_wr    = (const float*)d_in[8];
  const float* sp_br    = (const float*)d_in[9];
  const float* sp_wi    = (const float*)d_in[10];
  const float* sp_bi    = (const float*)d_in[11];
  const float* gate_w1  = (const float*)d_in[12];
  const float* gate_b1  = (const float*)d_in[13];
  const float* gate_w2  = (const float*)d_in[14];
  const float* gate_b2  = (const float*)d_in[15];
  const float* skip_w   = (const float*)d_in[16];
  const float* skip_b   = (const float*)d_in[17];
  const float* norm_g   = (const float*)d_in[18];
  const float* norm_b   = (const float*)d_in[19];
  const float* mlp_w1   = (const float*)d_in[20];
  const float* mlp_b1   = (const float*)d_in[21];
  const float* mlp_w2   = (const float*)d_in[22];
  const float* mlp_b2   = (const float*)d_in[23];
  float* out = (float*)d_out;

  float* W = (float*)d_ws;
  size_t off = 0;
  float* proj   = W + off; off += 640000;    // 10000 x 64
  float* spat   = W + off; off += 8388608;   // B,C,H,W
  float* mhf    = W + off; off += 8388608;
  float* spec   = W + off; off += 8388608;
  float* lo_r   = W + off; off += 278528;    // B,C,32,17
  float* lo_i   = W + off; off += 278528;
  float* ofr    = W + off; off += 278528;
  float* ofi    = W + off; off += 278528;
  float* spre   = W + off; off += 278528;
  float* spim   = W + off; off += 278528;
  float* m_sp   = W + off; off += 512;
  float* m_mhf  = W + off; off += 512;
  float* m_spec = W + off; off += 512;
  float* magb   = W + off; off += 8;
  float* gwb    = W + off; off += 24;
  float* bstat  = W + off; off += 16;
  int*   fidxb  = (int*)(W + off); off += 8;

  k_init<<<1, 32, 0, stream>>>(magb, bstat);
  k_se_proj<<<625, 128, 0, stream>>>(se_emb, se_w, se_b, proj);
  k_hash<<<512, 256, 0, stream>>>(x, proj, spat);
  k_lo<<<512, 256, 0, stream>>>(x, lo_r, lo_i, magb);
  k_einsum<<<544, 512, 0, stream>>>(lo_r, lo_i, mhf_wr, mhf_wi, ofr, ofi);
  k_fidx<<<1, 8, 0, stream>>>(magb, fidxb);
  k_specfreq<<<1088, 256, 0, stream>>>(fidxb, sp_emb, sp_wr, sp_br, sp_wi, sp_bi, spre, spim);
  k_irfft<<<512, 256, 0, stream>>>(ofr, ofi, mhf, mhf_bias, 1);
  k_irfft<<<512, 256, 0, stream>>>(spre, spim, spec, (const float*)nullptr, 0);
  k_mean<<<512, 256, 0, stream>>>(spat, m_sp);
  k_mean<<<512, 256, 0, stream>>>(mhf, m_mhf);
  k_mean<<<512, 256, 0, stream>>>(spec, m_spec);
  k_gate<<<1, 512, 0, stream>>>(m_sp, m_mhf, m_spec, gate_w1, gate_b1, gate_w2, gate_b2, gwb);
  k_y<<<1024, 256, 0, stream>>>(x, skip_w, skip_b, spat, mhf, spec, gwb, out, bstat);
  k_norm_mlp<<<1024, 256, 0, stream>>>(out, bstat, norm_g, norm_b, mlp_w1, mlp_b1, mlp_w2, mlp_b2);
}
